// GCONV_58059367907460
// MI455X (gfx1250) — matrix operand order repaired and checked by reading, not compiled, not run on hardware
//
#include <hip/hip_runtime.h>
#include <math.h>

typedef _Float16 v16h __attribute__((ext_vector_type(16)));
typedef _Float16 v8h  __attribute__((ext_vector_type(8)));
typedef float    v8f  __attribute__((ext_vector_type(8)));

#define NN 50000          // nodes
#define NE 300000         // edges
#define D  256            // feature dim
#define DCAT 512          // [aggr | x]
#define NG 512            // graphs
#define DH4 1024          // MLP hidden
#define DOUT 32

// ---------------- utility kernels ----------------

__global__ void k_zero_f32(float* __restrict__ p, long n) {
    long i = (long)blockIdx.x * blockDim.x + threadIdx.x;
    long s = (long)gridDim.x * blockDim.x;
    for (; i < n; i += s) p[i] = 0.f;
}

// edge scatter: aggr[dst] += x[src]; 64 threads per edge, float4 per thread
__global__ void k_scatter_add(const float* __restrict__ x,
                              const int* __restrict__ src,
                              const int* __restrict__ dst,
                              float* __restrict__ aggr) {
    long gid = (long)blockIdx.x * blockDim.x + threadIdx.x;
    long e = gid >> 6;
    int  c = (int)(gid & 63) << 2;
    if (e >= NE) return;
    int s = src[e], d = dst[e];
    const float4 v = *(const float4*)(x + (long)s * D + c);
    float* o = aggr + (long)d * D + c;
    atomicAdd(o + 0, v.x); atomicAdd(o + 1, v.y);
    atomicAdd(o + 2, v.z); atomicAdd(o + 3, v.w);
}

// cat[n][k] = f16( k<256 ? aggr[n][k] : x[n][k-256] )
__global__ void k_pack_cat(const float* __restrict__ aggr,
                           const float* __restrict__ x,
                           _Float16* __restrict__ cat) {
    long i = (long)blockIdx.x * blockDim.x + threadIdx.x;  // over NN*DCAT
    if (i >= (long)NN * DCAT) return;
    int  k = (int)(i & (DCAT - 1));
    long n = i >> 9;
    float v = (k < D) ? aggr[n * D + k] : x[n * D + (k - D)];
    cat[i] = (_Float16)v;
}

__global__ void k_cvt_f16(const float* __restrict__ in, _Float16* __restrict__ out, long n) {
    long i = (long)blockIdx.x * blockDim.x + threadIdx.x;
    if (i < n) out[i] = (_Float16)in[i];
}

// Wcat[l][n][k] (f16, NxK row-major) from W_rel[l][k][n], W_root[l][k-256][n]
__global__ void k_prep_wcat(const float* __restrict__ Wrel,
                            const float* __restrict__ Wroot,
                            _Float16* __restrict__ Wcat) {
    long i = (long)blockIdx.x * blockDim.x + threadIdx.x;  // 3*256*512
    if (i >= 3L * D * DCAT) return;
    int  k = (int)(i & (DCAT - 1));
    long r = i >> 9;
    int  n = (int)(r & (D - 1));
    int  l = (int)(r >> 8);
    float v = (k < D) ? Wrel [(long)l * D * D + (long)k * D + n]
                      : Wroot[(long)l * D * D + (long)(k - D) * D + n];
    Wcat[i] = (_Float16)v;
}

// W1t[n][k] (1024x256) from W1[k][n] (256x1024)
__global__ void k_prep_w1t(const float* __restrict__ W1, _Float16* __restrict__ W1t) {
    long i = (long)blockIdx.x * blockDim.x + threadIdx.x;  // 1024*256
    if (i >= (long)DH4 * D) return;
    int k = (int)(i & (D - 1));
    int n = (int)(i >> 8);
    W1t[i] = (_Float16)W1[(long)k * DH4 + n];
}

// W2t[n][k] (32x1024) from W2[k][n] (1024x32)
__global__ void k_prep_w2t(const float* __restrict__ W2, _Float16* __restrict__ W2t) {
    long i = (long)blockIdx.x * blockDim.x + threadIdx.x;  // 32*1024
    if (i >= (long)DOUT * DH4) return;
    int k = (int)(i & (DH4 - 1));
    int n = (int)(i >> 10);
    W2t[i] = (_Float16)W2[(long)k * DOUT + n];
}

// pooled[batch[n]] += x[n]
__global__ void k_pool(const float* __restrict__ x,
                       const int* __restrict__ batch,
                       float* __restrict__ pooled) {
    long gid = (long)blockIdx.x * blockDim.x + threadIdx.x;
    long n = gid >> 6;
    int  c = (int)(gid & 63) << 2;
    if (n >= NN) return;
    int g = batch[n];
    const float4 v = *(const float4*)(x + n * D + c);
    float* o = pooled + (long)g * D + c;
    atomicAdd(o + 0, v.x); atomicAdd(o + 1, v.y);
    atomicAdd(o + 2, v.z); atomicAdd(o + 3, v.w);
}

// ---------------- WMMA GEMM ----------------
// out[M][N] = act( A[M][K](f16) * Wt[N][K]^T (f16) + bias[N] )
// act: 0 = relu -> f32 out, 1 = relu -> f16 out, 2 = sigmoid -> f32 out
// One 16x16 tile per wave, 8 waves per block. K multiple of 32, M,N multiples of 16.
__global__ void k_wmma_gemm(const _Float16* __restrict__ A,
                            const _Float16* __restrict__ Wt,
                            const float* __restrict__ bias,
                            float* __restrict__ outF,
                            _Float16* __restrict__ outH,
                            int M, int K, int N, int act) {
    const int lane = threadIdx.x & 31;
    const int wave = threadIdx.x >> 5;
    const int tn = N >> 4;
    const int tiles = (M >> 4) * tn;
    const int t = blockIdx.x * (blockDim.x >> 5) + wave;
    if (t >= tiles) return;                       // wave-uniform exit (EXEC stays all-ones)
    const int m0 = (t / tn) << 4;
    const int n0 = (t - (t / tn) * tn) << 4;
    const int hi  = lane >> 4;                    // half-wave selector
    const int l15 = lane & 15;

    v8f c = {0.f, 0.f, 0.f, 0.f, 0.f, 0.f, 0.f, 0.f};
    const _Float16* arow = A  + (long)(m0 + l15) * K;   // A row for M = m0+l15
    const _Float16* brow = Wt + (long)(n0 + l15) * K;   // W^T row for N = n0+l15

    union Frag { v16h v; v8h h[2]; };
    for (int k0 = 0; k0 < K; k0 += 32) {
        Frag a, b;
        // A 16x32: elems 0-7 -> K = k0 + hi*8 .. +7 ; elems 8-15 -> K = k0+16+hi*8 .. +7
        a.h[0] = *(const v8h*)(arow + k0 + hi * 8);
        a.h[1] = *(const v8h*)(arow + k0 + 16 + hi * 8);
        // B 32x16 (as W^T rows): elems 0-7 -> K = k0 + hi*8 .. +7 ; elems 8-15 -> K = k0+16+hi*8 .. +7
        b.h[0] = *(const v8h*)(brow + k0 + hi * 8);
        b.h[1] = *(const v8h*)(brow + k0 + 16 + hi * 8);
        c = __builtin_amdgcn_wmma_f32_16x16x32_f16(
                false, a.v, false, b.v, (short)0, c, false, false);
    }

    const float bb = bias[n0 + l15];
#pragma unroll
    for (int r = 0; r < 8; ++r) {
        const int m = m0 + hi * 8 + r;            // C/D: VGPR r -> row hi*8+r
        float v = c[r] + bb;
        if (act == 2) v = 1.f / (1.f + __expf(-v));
        else          v = v > 0.f ? v : 0.f;
        const long o = (long)m * N + n0 + l15;
        if (act == 1) outH[o] = (_Float16)v;
        else          outF[o] = v;
    }
}

// ---------------- launch ----------------

extern "C" void kernel_launch(void* const* d_in, const int* in_sizes, int n_in,
                              void* d_out, int out_size, void* d_ws, size_t ws_size,
                              hipStream_t stream) {
    const float* d_x     = (const float*)d_in[0];
    const int*   d_edge  = (const int*)d_in[1];   // [2][NE]: src then dst
    const int*   d_batch = (const int*)d_in[2];
    const float* d_Wrel  = (const float*)d_in[3];
    const float* d_Wroot = (const float*)d_in[4];
    const float* d_bconv = (const float*)d_in[5];
    const float* d_W1    = (const float*)d_in[6];
    const float* d_b1    = (const float*)d_in[7];
    const float* d_W2    = (const float*)d_in[8];
    const float* d_b2    = (const float*)d_in[9];
    float* out = (float*)d_out;

    // workspace carve (256B aligned)
    char* w = (char*)d_ws;
    auto carve = [&](size_t bytes) { char* p = w; w += (bytes + 255) & ~(size_t)255; return p; };
    float*    aggr    = (float*)   carve((size_t)NN * D * 4);
    float*    y0      = (float*)   carve((size_t)NN * D * 4);
    float*    y1      = (float*)   carve((size_t)NN * D * 4);
    _Float16* cat     = (_Float16*)carve((size_t)NN * DCAT * 2);
    _Float16* Wcat    = (_Float16*)carve((size_t)3 * D * DCAT * 2);
    _Float16* W1t     = (_Float16*)carve((size_t)DH4 * D * 2);
    _Float16* W2t     = (_Float16*)carve((size_t)DOUT * DH4 * 2);
    float*    pooled  = (float*)   carve((size_t)NG * D * 4);
    _Float16* pooledH = (_Float16*)carve((size_t)NG * D * 2);
    _Float16* hH      = (_Float16*)carve((size_t)NG * DH4 * 2);

    // weight prep (f16 transposed copies), cheap, re-done every call
    k_prep_wcat<<<(3 * D * DCAT) / 256, 256, 0, stream>>>(d_Wrel, d_Wroot, Wcat);
    k_prep_w1t <<<(DH4 * D) / 256,      256, 0, stream>>>(d_W1, W1t);
    k_prep_w2t <<<(DOUT * DH4) / 256,   256, 0, stream>>>(d_W2, W2t);

    const float* xcur = d_x;
    float* ybuf[2] = {y0, y1};
    for (int l = 0; l < 3; ++l) {
        k_zero_f32<<<4096, 256, 0, stream>>>(aggr, (long)NN * D);
        k_scatter_add<<<(NE * 64) / 256, 256, 0, stream>>>(xcur, d_edge, d_edge + NE, aggr);
        k_pack_cat<<<((long)NN * DCAT) / 256, 256, 0, stream>>>(aggr, xcur, cat);
        float* yout = ybuf[l & 1];
        // M=50000, K=512, N=256 -> 50000 tiles / 8 waves = 6250 blocks
        k_wmma_gemm<<<6250, 256, 0, stream>>>(cat, Wcat + (long)l * D * DCAT,
                                              d_bconv + l * D, yout, nullptr,
                                              NN, DCAT, D, 0);
        xcur = yout;
    }

    // global add pool
    k_zero_f32<<<512, 256, 0, stream>>>(pooled, (long)NG * D);
    k_pool<<<(NN * 64) / 256, 256, 0, stream>>>(xcur, d_batch, pooled);
    k_cvt_f16<<<(NG * D) / 256, 256, 0, stream>>>(pooled, pooledH, (long)NG * D);

    // MLP: h = relu(pooled @ W1 + b1) -> f16 ; out = sigmoid(h @ W2 + b2) -> f32
    // GEMM1: 512x256 * 256x1024 -> 2048 tiles / 8 = 256 blocks
    k_wmma_gemm<<<256, 256, 0, stream>>>(pooledH, W1t, d_b1, nullptr, hH,
                                         NG, D, DH4, 1);
    // GEMM2: 512x1024 * 1024x32 -> 64 tiles / 8 = 8 blocks
    k_wmma_gemm<<<8, 256, 0, stream>>>(hH, W2t, d_b2, out, nullptr,
                                       NG, DH4, DOUT, 2);
    (void)in_sizes; (void)n_in; (void)out_size; (void)ws_size;
}
